// seq2seq_Decoder_3298534883635
// MI455X (gfx1250) — matrix operand order repaired and checked by reading, not compiled, not run on hardware
//
#include <hip/hip_runtime.h>
#include <hip/hip_bf16.h>
#include <math.h>

typedef __bf16 bf16_t;
typedef __attribute__((ext_vector_type(16))) __bf16 v16bf;
typedef __attribute__((ext_vector_type(8)))  float  v8f;

#define B_   32
#define T_   64
#define H_   1024
#define E_   512
#define V_   32000
#define EH_  1536
#define G3_  3072
#define NWG_ 16
#define HS_  1032   // padded LDS stride for h-state rows (conflict-free, 16B aligned)
#define HS4_ 129    // HS_ in uint4 units

// ---------------------------------------------------------------------------
// WMMA helpers (layouts per CDNA5 ISA 7.12.2)
// ---------------------------------------------------------------------------
__device__ inline v8f wmma_bf16(v16bf a, v16bf b, v8f c) {
  return __builtin_amdgcn_wmma_f32_16x16x32_bf16(false, a, false, b, (short)0, c,
                                                 false, false);
}

union frag_u { uint4 u[2]; v16bf v; };

// A: 16x32 bf16 tile, row-major, stride lda elements. Requires base and
// row stride 16B-aligned -> two b128 loads per lane.
__device__ inline v16bf load_A_frag(const bf16_t* A, int lda, int lane) {
  const int row = lane & 15;
  const int kb  = (lane & 16) ? 16 : 0;
  const bf16_t* r = A + row * lda + kb;
  frag_u u;
  u.u[0] = *(const uint4*)(r);
  u.u[1] = *(const uint4*)(r + 8);
  return u.v;
}

// B fragment from "fragment-major" swizzled storage: 32 contiguous bytes/lane.
__device__ inline v16bf load_B_swz(const bf16_t* base, int lane) {
  const bf16_t* p = base + (lane << 4);
  frag_u u;
  u.u[0] = *(const uint4*)(p);
  u.u[1] = *(const uint4*)(p + 8);
  return u.v;
}

__device__ inline float sigmoidf_(float x) { return 1.0f / (1.0f + __expf(-x)); }

// ---------------------------------------------------------------------------
// Prep: embedding + concat(context) -> dec_in bf16 [T][B][E+H]
// ---------------------------------------------------------------------------
__global__ __launch_bounds__(256) void k_embed(const int* __restrict__ X,
                                               const float* __restrict__ emb,
                                               const float* __restrict__ h0,
                                               bf16_t* __restrict__ dec_in) {
  int idx = blockIdx.x * 256 + threadIdx.x;   // T_*B_*EH_ total
  int e   = idx % EH_;
  int row = idx / EH_;                        // row = t*B + b
  int t = row >> 5, b = row & 31;
  float v;
  if (e < E_) {
    int x = X[b * T_ + t];
    v = emb[x * E_ + e];
  } else {
    v = h0[(B_ + b) * H_ + (e - E_)];         // context = h0[-1]
  }
  dec_in[idx] = (bf16_t)v;
}

// Convert [1024][3072] fp32 weight into WMMA-B fragment-major bf16 layout:
// dst[ kt(32) ][ nt(192) ][ lane(32) ][ h(16) ],
// where k = kt*32 + (lane&16) + h, n = nt*16 + (lane&15).
__global__ __launch_bounds__(256) void k_cvt_swz(const float* __restrict__ src,
                                                 bf16_t* __restrict__ dst) {
  int o    = blockIdx.x * 256 + threadIdx.x;  // H_*G3_ total
  int h    = o & 15;
  int lane = (o >> 4) & 31;
  int nt   = (o >> 9) % 192;
  int kt   = o / (512 * 192);
  int k    = kt * 32 + (lane & 16) + h;
  int n    = nt * 16 + (lane & 15);
  dst[o] = (bf16_t)src[k * G3_ + n];
}

// ---------------------------------------------------------------------------
// Tiled GEMM body shared by gx0 & logits: C[64x64] = A(bf16) x B(fp32->bf16)
// A tile LDS: [64][40] (b128 stores/loads, conflict-free)
// B tile LDS: swizzled [4 nchunk][32 lane][16 h]
// ---------------------------------------------------------------------------
__device__ inline void gemm_tile(const bf16_t* __restrict__ A, int lda,
                                 const float* __restrict__ W, int ldw, int K,
                                 int m0, int n0, bf16_t (*As)[40],
                                 bf16_t* BsSw, int tid, v8f& acc0, v8f& acc1) {
  const int w = tid >> 5, lane = tid & 31;
  const int mrow = (w & 3) * 16, nc = (w >> 2) * 32;
  const int nchunk0 = nc >> 4;
  for (int k = 0; k < K; k += 32) {
    {  // stage A 64x32 bf16 : one b128 load + one b128 LDS store per thread
      int r = tid >> 2, c = (tid & 3) * 8;
      uint4 v = *(const uint4*)(A + (m0 + r) * lda + k + c);
      *(uint4*)(&As[r][c]) = v;
    }
    {  // stage B 32x64 fp32 -> bf16 swizzled
      int r = tid >> 3, c = (tid & 7) * 8;
      const float* Wp = W + (k + r) * ldw + n0 + c;
      float4 v0 = *(const float4*)(Wp);
      float4 v1 = *(const float4*)(Wp + 4);
      int base = ((((c >> 4) * 32) + (r & 16) + (c & 15)) << 4) + (r & 15);
      BsSw[base + 0 * 16] = (bf16_t)v0.x;
      BsSw[base + 1 * 16] = (bf16_t)v0.y;
      BsSw[base + 2 * 16] = (bf16_t)v0.z;
      BsSw[base + 3 * 16] = (bf16_t)v0.w;
      BsSw[base + 4 * 16] = (bf16_t)v1.x;
      BsSw[base + 5 * 16] = (bf16_t)v1.y;
      BsSw[base + 6 * 16] = (bf16_t)v1.z;
      BsSw[base + 7 * 16] = (bf16_t)v1.w;
    }
    __syncthreads();
    v16bf a   = load_A_frag(&As[mrow][0], 40, lane);
    v16bf bf0 = load_B_swz(BsSw + ((nchunk0 + 0) << 9), lane);
    v16bf bf1 = load_B_swz(BsSw + ((nchunk0 + 1) << 9), lane);
    acc0 = wmma_bf16(a, bf0, acc0);
    acc1 = wmma_bf16(a, bf1, acc1);
    __syncthreads();
  }
}

// Gx0 = dec_in @ W0 + b0[0] : M=2048, K=1536, N=3072 (fp32 out)
__global__ __launch_bounds__(256) void k_gemm_gx0(const bf16_t* __restrict__ A,
                                                  const float* __restrict__ W0,
                                                  const float* __restrict__ b0,
                                                  float* __restrict__ Gx0) {
  __shared__ bf16_t As[64][40];
  __shared__ bf16_t BsSw[4 * 32 * 16];
  const int m0 = blockIdx.x * 64, n0 = blockIdx.y * 64;
  const int tid = threadIdx.x, w = tid >> 5, lane = tid & 31;
  const int mrow = (w & 3) * 16, nc = (w >> 2) * 32;
  v8f acc0 = {}, acc1 = {};
  gemm_tile(A, EH_, W0, G3_, EH_, m0, n0, As, BsSw, tid, acc0, acc1);
  const int col = lane & 15, rb = (lane & 16) ? 8 : 0;
  const int gc0 = n0 + nc + col, gc1 = gc0 + 16;
  const float bb0 = b0[gc0], bb1 = b0[gc1];
#pragma unroll
  for (int i = 0; i < 8; ++i) {
    int row = m0 + mrow + rb + i;
    Gx0[row * G3_ + gc0] = acc0[i] + bb0;
    Gx0[row * G3_ + gc1] = acc1[i] + bb1;
  }
}

// logits = seq @ Wd + bd : M=2048, K=1024, N=32000 (fp32 out, HBM-store bound)
__global__ __launch_bounds__(256) void k_logits(const bf16_t* __restrict__ A,
                                                const float* __restrict__ Wd,
                                                const float* __restrict__ bd,
                                                float* __restrict__ out) {
  __shared__ bf16_t As[64][40];
  __shared__ bf16_t BsSw[4 * 32 * 16];
  const int m0 = blockIdx.x * 64, n0 = blockIdx.y * 64;
  const int tid = threadIdx.x, w = tid >> 5, lane = tid & 31;
  const int mrow = (w & 3) * 16, nc = (w >> 2) * 32;
  v8f acc0 = {}, acc1 = {};
  gemm_tile(A, H_, Wd, V_, H_, m0, n0, As, BsSw, tid, acc0, acc1);
  const int col = lane & 15, rb = (lane & 16) ? 8 : 0;
  const int gc0 = n0 + nc + col, gc1 = gc0 + 16;
  const float bb0 = bd[gc0], bb1 = bd[gc1];
#pragma unroll
  for (int i = 0; i < 8; ++i) {
    int row = m0 + mrow + rb + i;
    out[row * V_ + gc0] = acc0[i] + bb0;
    out[row * V_ + gc1] = acc1[i] + bb1;
  }
}

// ---------------------------------------------------------------------------
// Persistent cooperative recurrence: 16 WGs x 256 thr, each owns 64 h-columns.
// ---------------------------------------------------------------------------
__device__ inline void grid_sync(unsigned* bar, unsigned target) {
  __threadfence();
  __syncthreads();
  if (threadIdx.x == 0) {
    atomicAdd(bar, 1u);
    while (__hip_atomic_load(bar, __ATOMIC_RELAXED, __HIP_MEMORY_SCOPE_AGENT) <
           target)
      __builtin_amdgcn_s_sleep(2);
  }
  __syncthreads();
  __threadfence();
}

__global__ __launch_bounds__(256) void k_recurrent(
    const float* __restrict__ Gx0, const bf16_t* __restrict__ U0s,
    const bf16_t* __restrict__ W1s, const bf16_t* __restrict__ U1s,
    const float* __restrict__ b0, const float* __restrict__ b1,
    const float* __restrict__ h0, bf16_t* __restrict__ h1b,
    bf16_t* __restrict__ h2b, bf16_t* __restrict__ seqb,
    float* __restrict__ out_states, unsigned* __restrict__ bar) {
  extern __shared__ char smem[];
  bf16_t* h1s = (bf16_t*)smem;             // [32][HS_] padded
  bf16_t* h2s = h1s + B_ * HS_;            // [32][HS_] padded

  const int tid  = threadIdx.x;
  const int w    = tid >> 5, lane = tid & 31;
  const int mrow = (w & 1) * 16;           // batch-row base of this wave's tile
  const int ncol = (w >> 1) * 16;          // col base inside 64-col slice
  const int j0   = blockIdx.x * 64;        // owned h-column slice
  const int col  = lane & 15;
  const int rb   = (lane & 16) ? 8 : 0;
  const int jg   = j0 + ncol + col;        // global h column for this lane
  const int ntb  = (j0 + ncol) >> 4;       // B ntile for this wave's slice

  // per-thread fp32 hidden state for the owned tile (exact z*h+(1-z)*hc)
  float h1r[8], h2r[8];
#pragma unroll
  for (int i = 0; i < 8; ++i) {
    int bg = mrow + rb + i;
    h1r[i] = h0[bg * H_ + jg];
    h2r[i] = h0[(B_ + bg) * H_ + jg];
  }
  // seed LDS copies of full h1/h2 (bf16, padded stride)
  for (int i = tid; i < B_ * H_; i += 256) {
    int b = i >> 10, j = i & (H_ - 1);
    h1s[b * HS_ + j] = (bf16_t)h0[i];
    h2s[b * HS_ + j] = (bf16_t)h0[B_ * H_ + i];
  }
  __syncthreads();

  unsigned epoch = 0;
  for (int t = 0; t < T_; ++t) {
    // ---------------- phase A: gh0 = h1 @ U0 (3 gate tiles) + gates --------
    v8f az = {}, ar = {}, ah = {};
    for (int k = 0; k < H_; k += 32) {
      const int kt = k >> 5;
      v16bf a  = load_A_frag(h1s + mrow * HS_ + k, HS_, lane);
      v16bf bz = load_B_swz(U0s + (((kt * 192) + 0 * 64 + ntb) << 9), lane);
      v16bf br = load_B_swz(U0s + (((kt * 192) + 1 * 64 + ntb) << 9), lane);
      v16bf bh = load_B_swz(U0s + (((kt * 192) + 2 * 64 + ntb) << 9), lane);
      az = wmma_bf16(a, bz, az);
      ar = wmma_bf16(a, br, ar);
      ah = wmma_bf16(a, bh, ah);
    }
    {
      const float bbz = b0[G3_ + 0 * H_ + jg];
      const float bbr = b0[G3_ + 1 * H_ + jg];
      const float bbh = b0[G3_ + 2 * H_ + jg];
#pragma unroll
      for (int i = 0; i < 8; ++i) {
        int bg = mrow + rb + i;
        const float* gx = Gx0 + (t * B_ + bg) * G3_;  // includes b0[0]
        float z  = sigmoidf_(gx[0 * H_ + jg] + az[i] + bbz);
        float r  = sigmoidf_(gx[1 * H_ + jg] + ar[i] + bbr);
        float hc = tanhf(gx[2 * H_ + jg] + r * (ah[i] + bbh));
        float hn = z * h1r[i] + (1.0f - z) * hc;
        h1r[i] = hn;
        h1b[bg * H_ + jg] = (bf16_t)hn;
        if (t == T_ - 1) out_states[bg * H_ + jg] = hn;
      }
    }
    grid_sync(bar, (++epoch) * NWG_);
    {  // refresh full h1 in LDS (b128 copies into padded rows)
      const uint4* s = (const uint4*)h1b;
      uint4* d = (uint4*)h1s;
      for (int i = tid; i < (B_ * H_) / 8; i += 256)
        d[(i >> 7) * HS4_ + (i & 127)] = s[i];
    }
    __syncthreads();

    // ---------------- phase B: gx1 = h1'@W1, gh1 = h2@U1 + gates -----------
    v8f xz = {}, xr = {}, xh = {}, hz = {}, hr = {}, hh = {};
    for (int k = 0; k < H_; k += 32) {
      const int kt = k >> 5;
      v16bf a1 = load_A_frag(h1s + mrow * HS_ + k, HS_, lane);
      v16bf a2 = load_A_frag(h2s + mrow * HS_ + k, HS_, lane);
      v16bf b;
      b  = load_B_swz(W1s + (((kt * 192) + 0 * 64 + ntb) << 9), lane);
      xz = wmma_bf16(a1, b, xz);
      b  = load_B_swz(W1s + (((kt * 192) + 1 * 64 + ntb) << 9), lane);
      xr = wmma_bf16(a1, b, xr);
      b  = load_B_swz(W1s + (((kt * 192) + 2 * 64 + ntb) << 9), lane);
      xh = wmma_bf16(a1, b, xh);
      b  = load_B_swz(U1s + (((kt * 192) + 0 * 64 + ntb) << 9), lane);
      hz = wmma_bf16(a2, b, hz);
      b  = load_B_swz(U1s + (((kt * 192) + 1 * 64 + ntb) << 9), lane);
      hr = wmma_bf16(a2, b, hr);
      b  = load_B_swz(U1s + (((kt * 192) + 2 * 64 + ntb) << 9), lane);
      hh = wmma_bf16(a2, b, hh);
    }
    {
      const float bxz = b1[0 * H_ + jg], bhz = b1[G3_ + 0 * H_ + jg];
      const float bxr = b1[1 * H_ + jg], bhr = b1[G3_ + 1 * H_ + jg];
      const float bxh = b1[2 * H_ + jg], bhh = b1[G3_ + 2 * H_ + jg];
#pragma unroll
      for (int i = 0; i < 8; ++i) {
        int bg = mrow + rb + i;
        float z  = sigmoidf_((xz[i] + bxz) + (hz[i] + bhz));
        float r  = sigmoidf_((xr[i] + bxr) + (hr[i] + bhr));
        float hc = tanhf((xh[i] + bxh) + r * (hh[i] + bhh));
        float hn = z * h2r[i] + (1.0f - z) * hc;
        h2r[i] = hn;
        h2b[bg * H_ + jg] = (bf16_t)hn;
        seqb[(bg * T_ + t) * H_ + jg] = (bf16_t)hn;   // row b*T+t for logits
        if (t == T_ - 1) out_states[B_ * H_ + bg * H_ + jg] = hn;
      }
    }
    grid_sync(bar, (++epoch) * NWG_);
    {  // refresh full h2 in LDS
      const uint4* s = (const uint4*)h2b;
      uint4* d = (uint4*)h2s;
      for (int i = tid; i < (B_ * H_) / 8; i += 256)
        d[(i >> 7) * HS4_ + (i & 127)] = s[i];
    }
    __syncthreads();
  }
}

// ---------------------------------------------------------------------------
// Host-side orchestration
// ---------------------------------------------------------------------------
extern "C" void kernel_launch(void* const* d_in, const int* in_sizes, int n_in,
                              void* d_out, int out_size, void* d_ws,
                              size_t ws_size, hipStream_t stream) {
  const int*   X   = (const int*)d_in[0];
  const float* h0  = (const float*)d_in[1];
  const float* emb = (const float*)d_in[2];
  const float* W0  = (const float*)d_in[3];
  const float* U0  = (const float*)d_in[4];
  const float* b0  = (const float*)d_in[5];
  const float* W1  = (const float*)d_in[6];
  const float* U1  = (const float*)d_in[7];
  const float* b1  = (const float*)d_in[8];
  const float* Wd  = (const float*)d_in[9];
  const float* bd  = (const float*)d_in[10];
  float* out = (float*)d_out;

  char* ws = (char*)d_ws;
  size_t off = 0;
  auto carve = [&](size_t bytes) {
    char* p = ws + off;
    off = (off + bytes + 255) & ~(size_t)255;
    return p;
  };
  bf16_t*   decin = (bf16_t*)carve((size_t)T_ * B_ * EH_ * 2);   // 6 MB
  bf16_t*   U0s   = (bf16_t*)carve((size_t)H_ * G3_ * 2);        // 6.3 MB
  bf16_t*   W1s   = (bf16_t*)carve((size_t)H_ * G3_ * 2);
  bf16_t*   U1s   = (bf16_t*)carve((size_t)H_ * G3_ * 2);
  float*    Gx0   = (float*)carve((size_t)T_ * B_ * G3_ * 4);    // 25 MB
  bf16_t*   seqb  = (bf16_t*)carve((size_t)B_ * T_ * H_ * 2);    // 4 MB
  bf16_t*   h1b   = (bf16_t*)carve((size_t)B_ * H_ * 2);
  bf16_t*   h2b   = (bf16_t*)carve((size_t)B_ * H_ * 2);
  unsigned* bar   = (unsigned*)carve(256);
  (void)ws_size; (void)in_sizes; (void)n_in; (void)out_size;

  hipMemsetAsync(bar, 0, 256, stream);

  k_embed<<<(T_ * B_ * EH_) / 256, 256, 0, stream>>>(X, emb, h0, decin);
  k_cvt_swz<<<(H_ * G3_) / 256, 256, 0, stream>>>(U0, U0s);
  k_cvt_swz<<<(H_ * G3_) / 256, 256, 0, stream>>>(W1, W1s);
  k_cvt_swz<<<(H_ * G3_) / 256, 256, 0, stream>>>(U1, U1s);

  k_gemm_gx0<<<dim3((T_ * B_) / 64, G3_ / 64), 256, 0, stream>>>(decin, W0, b0,
                                                                 Gx0);

  k_recurrent<<<NWG_, 256, 2 * B_ * HS_ * sizeof(bf16_t), stream>>>(
      Gx0, U0s, W1s, U1s, b0, b1, h0, h1b, h2b, seqb,
      out + (size_t)B_ * T_ * V_, bar);

  k_logits<<<dim3((B_ * T_) / 64, V_ / 64), 256, 0, stream>>>(seqb, Wd, bd,
                                                              out);
}
